// CausalSelfAttention_16381005267609
// MI455X (gfx1250) — matrix operand order repaired and checked by reading, not compiled, not run on hardware
//
#include <hip/hip_runtime.h>
#include <stdint.h>
#include <stddef.h>

#define D_MODEL 1024
#define N_HEADS 16
#define D_HEAD  64
#define BATCH   4
#define SEQ     2048
#define BT      (BATCH*SEQ)      /* 8192 rows */
#define QKV_N   (3*D_MODEL)      /* 3072 */
#define LDSPAD  40               /* 32 keys + 8 pad bf16 -> 80B row stride, bank-friendly */

typedef __bf16 bf16_t;
typedef __bf16 bf16x16 __attribute__((ext_vector_type(16)));
typedef __bf16 bf16x8  __attribute__((ext_vector_type(8)));
typedef float  f32x8   __attribute__((ext_vector_type(8)));

union Frag { bf16x16 v; bf16x8 h[2]; };

__device__ __forceinline__ bf16_t f2bf(float f) {
  uint32_t u = __float_as_uint(f);
  uint32_t r = u + 0x7FFFu + ((u >> 16) & 1u);   // round-to-nearest-even
  uint16_t s = (uint16_t)(r >> 16);
  return __builtin_bit_cast(bf16_t, s);
}

__device__ __forceinline__ f32x8 wmma_bf16(const Frag& a, const Frag& b, f32x8 c) {
  return __builtin_amdgcn_wmma_f32_16x16x32_bf16(false, a.v, false, b.v,
                                                 (short)0, c, false, false);
}

/* ---------------- stage 0: precision conversion ---------------- */

__global__ void k_f32_to_bf16(const float* __restrict__ in, bf16_t* __restrict__ out, int n) {
  int i = blockIdx.x * blockDim.x + threadIdx.x;
  if (i < n) out[i] = f2bf(in[i]);
}

/* w is [O][I] (torch Linear); produce wT [I][O] in bf16.
   32x32 LDS-tiled transpose: coalesced reads AND writes. Block = 256 = 32x8. */
__global__ void __launch_bounds__(256)
k_transpose_to_bf16(const float* __restrict__ w, bf16_t* __restrict__ wT, int O, int I) {
  __shared__ float tile[32][33];                 /* +1 pad: no bank conflicts */
  int tx = threadIdx.x & 31;
  int ty = threadIdx.x >> 5;                     /* 0..7 */
  int to = blockIdx.x % (O / 32);
  int ti = blockIdx.x / (O / 32);
  int o0 = to * 32, i0 = ti * 32;
  #pragma unroll
  for (int r = 0; r < 4; ++r)                    /* read w rows, coalesced along I */
    tile[ty + 8 * r][tx] = w[(size_t)(o0 + ty + 8 * r) * I + i0 + tx];
  __syncthreads();
  #pragma unroll
  for (int r = 0; r < 4; ++r)                    /* write wT rows, coalesced along O */
    wT[(size_t)(i0 + ty + 8 * r) * O + o0 + tx] = f2bf(tile[tx][ty + 8 * r]);
}

/* ---------------- shared 32x64-tile GEMM core (C = X[M,1024] @ WT[1024,LDB] tile) -------- */

template<int LDB>
__device__ __forceinline__ void gemm_core(const bf16_t* __restrict__ X,
                                          const bf16_t* __restrict__ WT,
                                          int m0, int n0, int lane,
                                          f32x8 acc[2][4]) {
  const int lrow = lane & 15;
  const int koff = (lane & 16) ? 8 : 0;   /* A-frag K half split per lane group */
  const bf16_t* a0 = X + (size_t)(m0 + lrow) * D_MODEL + koff;
  const bf16_t* a1 = a0 + (size_t)16 * D_MODEL;
  const bf16_t* br = WT + (size_t)koff * LDB + n0 + lrow;   /* B-frag: lane = n col */
  #pragma unroll 2
  for (int kk = 0; kk < D_MODEL; kk += 32) {
    /* pull next K-step toward the WGP while this step's WMMAs drain */
    if (kk + 32 < D_MODEL) {
      __builtin_prefetch(a0 + kk + 32, 0, 3);
      __builtin_prefetch(a1 + kk + 32, 0, 3);
      __builtin_prefetch(br + (size_t)(kk + 32) * LDB, 0, 3);
    }
    Frag a[2], b[4];
    a[0].h[0] = *(const bf16x8*)(a0 + kk);
    a[0].h[1] = *(const bf16x8*)(a0 + kk + 16);
    a[1].h[0] = *(const bf16x8*)(a1 + kk);
    a[1].h[1] = *(const bf16x8*)(a1 + kk + 16);
    #pragma unroll
    for (int j = 0; j < 4; ++j) {
      const bf16_t* bp = br + (size_t)kk * LDB + j * 16;
      #pragma unroll
      for (int s = 0; s < 8; ++s) {
        b[j].h[0][s] = bp[(size_t)s * LDB];
        b[j].h[1][s] = bp[(size_t)(16 + s) * LDB];
      }
    }
    #pragma unroll
    for (int i = 0; i < 2; ++i)
      #pragma unroll
      for (int j = 0; j < 4; ++j)
        acc[i][j] = wmma_bf16(a[i], b[j], acc[i][j]);
  }
}

/* ---------------- stage 1: QKV projection, scatter to Q (scaled), K^T, V --------------- */

__global__ void __launch_bounds__(256)
k_qkv_gemm(const bf16_t* __restrict__ X,     /* [BT][1024] bf16 */
           const bf16_t* __restrict__ WT,    /* [1024][3072] bf16 */
           bf16_t* __restrict__ Q,           /* [B,H,T,64], pre-scaled by 1/8 */
           bf16_t* __restrict__ KT,          /* [B,H,64,T] */
           bf16_t* __restrict__ V) {         /* [B,H,T,64] */
  const int NT = QKV_N / 64;                 /* 48 col tiles */
  int wid  = blockIdx.x * (blockDim.x >> 5) + (threadIdx.x >> 5);
  int m0   = (wid / NT) * 32;
  int n0   = (wid % NT) * 64;
  int lane = threadIdx.x & 31;
  int lrow = lane & 15;
  int rsh  = (lane & 16) ? 8 : 0;

  f32x8 acc[2][4];
  #pragma unroll
  for (int i = 0; i < 2; ++i)
    #pragma unroll
    for (int j = 0; j < 4; ++j)
      acc[i][j] = (f32x8){0.f,0.f,0.f,0.f,0.f,0.f,0.f,0.f};

  gemm_core<QKV_N>(X, WT, m0, n0, lane, acc);

  #pragma unroll
  for (int i = 0; i < 2; ++i) {
    #pragma unroll
    for (int r = 0; r < 8; ++r) {
      int row = m0 + i * 16 + r + rsh;       /* C layout: lanes16-31 own M=8..15 */
      int b_  = row / SEQ, t = row % SEQ;
      #pragma unroll
      for (int j = 0; j < 4; ++j) {
        int o = n0 + j * 16 + lrow;          /* output feature */
        float val = acc[i][j][r];
        int s = o >> 10;                     /* 0=q 1=k 2=v (uniform per tile) */
        int h = (o & 1023) >> 6;
        int d = o & 63;
        size_t bh = (size_t)(b_ * N_HEADS + h);
        if (s == 0)      Q [(bh * SEQ + t) * D_HEAD + d] = f2bf(val * 0.125f);
        else if (s == 1) KT[(bh * D_HEAD + d) * SEQ + t] = f2bf(val);
        else             V [(bh * SEQ + t) * D_HEAD + d] = f2bf(val);
      }
    }
  }
}

/* ---------------- stage 2: flash causal attention, 1 wave = (b,h) x 16 queries --------- */

__global__ void __launch_bounds__(256)
k_attn(const bf16_t* __restrict__ Q,   /* [B,H,T,64] scaled */
       const bf16_t* __restrict__ KT,  /* [B,H,64,T] */
       const bf16_t* __restrict__ V,   /* [B,H,T,64] */
       bf16_t* __restrict__ Y) {       /* [B,T,C] */
  __shared__ bf16_t lds[8][16 * LDSPAD];
  int wslot = threadIdx.x >> 5;
  bf16_t* P = lds[wslot];

  int wid  = blockIdx.x * (blockDim.x >> 5) + wslot;
  int bh   = wid >> 7;                 /* 64 (b,h) pairs */
  int q0   = (wid & 127) * 16;
  int lane = threadIdx.x & 31;
  int lrow = lane & 15;
  int khalf = (lane >> 4) & 1;
  int koff  = khalf * 8;

  const bf16_t* Qb = Q  + ((size_t)bh * SEQ + q0) * D_HEAD;
  const bf16_t* Kb = KT + (size_t)bh * D_HEAD * SEQ;
  const bf16_t* Vb = V  + (size_t)bh * SEQ * D_HEAD;

  Frag qa[2];                          /* Q tile 16x64 as two K=32 A-frags */
  #pragma unroll
  for (int kf = 0; kf < 2; ++kf) {
    const bf16_t* p = Qb + (size_t)lrow * D_HEAD + kf * 32 + koff;
    qa[kf].h[0] = *(const bf16x8*)(p);
    qa[kf].h[1] = *(const bf16x8*)(p + 16);
  }

  f32x8 o[4];
  #pragma unroll
  for (int j = 0; j < 4; ++j) o[j] = (f32x8){0.f,0.f,0.f,0.f,0.f,0.f,0.f,0.f};
  float m[8], l[8];
  #pragma unroll
  for (int r = 0; r < 8; ++r) { m[r] = -1e30f; l[r] = 0.f; }

  const int kend = q0 + 16;
  for (int kbase = 0; kbase < kend; kbase += 32) {
    if (kbase + 32 < kend) {           /* prefetch next chunk's K rows and V rows */
      __builtin_prefetch(Kb + (size_t)lane * SEQ + kbase + 32, 0, 3);
      __builtin_prefetch(Kb + (size_t)(32 + lane) * SEQ + kbase + 32, 0, 3);
      __builtin_prefetch(Vb + (size_t)(kbase + 32 + lane) * D_HEAD, 0, 3);
    }
    f32x8 s0 = (f32x8){0.f,0.f,0.f,0.f,0.f,0.f,0.f,0.f};
    f32x8 s1 = s0;
    #pragma unroll
    for (int kf = 0; kf < 2; ++kf) {   /* S = Q K^T over Dh */
      Frag kb0, kb1;
      const bf16_t* p = Kb + (size_t)(kf * 32 + koff) * SEQ + kbase + lrow;
      #pragma unroll
      for (int s = 0; s < 8; ++s) {
        kb0.h[0][s] = p[(size_t)s * SEQ];      kb0.h[1][s] = p[(size_t)(16 + s) * SEQ];
        kb1.h[0][s] = p[(size_t)s * SEQ + 16]; kb1.h[1][s] = p[(size_t)(16 + s) * SEQ + 16];
      }
      s0 = wmma_bf16(qa[kf], kb0, s0);
      s1 = wmma_bf16(qa[kf], kb1, s1);
    }
    if (kbase + 31 >= q0) {            /* causal mask on diagonal-crossing chunks */
      #pragma unroll
      for (int r = 0; r < 8; ++r) {
        int row  = q0 + r + khalf * 8;
        int key0 = kbase + lrow;
        if (key0 > row)      s0[r] = -1e30f;
        if (key0 + 16 > row) s1[r] = -1e30f;
      }
    }
    float alpha[8];
    #pragma unroll
    for (int r = 0; r < 8; ++r) {      /* online softmax; each 16-lane half owns rows */
      float mx = fmaxf(s0[r], s1[r]);
      mx = fmaxf(mx, __shfl_xor(mx, 1, 32));
      mx = fmaxf(mx, __shfl_xor(mx, 2, 32));
      mx = fmaxf(mx, __shfl_xor(mx, 4, 32));
      mx = fmaxf(mx, __shfl_xor(mx, 8, 32));
      float mn = fmaxf(m[r], mx);
      alpha[r] = __expf(m[r] - mn);
      m[r] = mn;
      float p0 = __expf(s0[r] - mn);
      float p1 = __expf(s1[r] - mn);
      s0[r] = p0; s1[r] = p1;
      float rs = p0 + p1;
      rs += __shfl_xor(rs, 1, 32);
      rs += __shfl_xor(rs, 2, 32);
      rs += __shfl_xor(rs, 4, 32);
      rs += __shfl_xor(rs, 8, 32);
      l[r] = l[r] * alpha[r] + rs;
    }
    #pragma unroll
    for (int j = 0; j < 4; ++j)
      #pragma unroll
      for (int r = 0; r < 8; ++r)
        o[j][r] *= alpha[r];

    /* C-layout P -> LDS -> A-layout frag (in-wave transpose) */
    #pragma unroll
    for (int r = 0; r < 8; ++r) {
      int row = r + khalf * 8;
      P[row * LDSPAD + lrow]      = f2bf(s0[r]);
      P[row * LDSPAD + 16 + lrow] = f2bf(s1[r]);
    }
    __builtin_amdgcn_wave_barrier();
    asm volatile("s_wait_dscnt 0" ::: "memory");
    Frag pa;
    {
      const bf16_t* p = P + lrow * LDSPAD + koff;
      pa.h[0] = *(const bf16x8*)(p);
      pa.h[1] = *(const bf16x8*)(p + 16);
    }
    __builtin_amdgcn_wave_barrier();

    #pragma unroll
    for (int j = 0; j < 4; ++j) {      /* O += P V */
      Frag vb;
      const bf16_t* p = Vb + (size_t)(kbase + koff) * D_HEAD + j * 16 + lrow;
      #pragma unroll
      for (int s = 0; s < 8; ++s) {
        vb.h[0][s] = p[(size_t)s * D_HEAD];
        vb.h[1][s] = p[(size_t)(16 + s) * D_HEAD];
      }
      o[j] = wmma_bf16(pa, vb, o[j]);
    }
  }

  int b_ = bh >> 4, h_ = bh & 15;
  #pragma unroll
  for (int r = 0; r < 8; ++r) {
    float inv = 1.0f / l[r];
    int row = q0 + r + khalf * 8;
    #pragma unroll
    for (int j = 0; j < 4; ++j)
      Y[((size_t)b_ * SEQ + row) * D_MODEL + h_ * D_HEAD + j * 16 + lrow] =
          f2bf(o[j][r] * inv);
  }
}

/* ---------------- stage 3: output projection -> fp32 ---------------- */

__global__ void __launch_bounds__(256)
k_out_gemm(const bf16_t* __restrict__ Y,    /* [BT][1024] bf16 */
           const bf16_t* __restrict__ WT,   /* [1024][1024] bf16 */
           float* __restrict__ OUT) {       /* [BT][1024] f32 */
  const int NT = D_MODEL / 64;              /* 16 col tiles */
  int wid  = blockIdx.x * (blockDim.x >> 5) + (threadIdx.x >> 5);
  int m0   = (wid / NT) * 32;
  int n0   = (wid % NT) * 64;
  int lane = threadIdx.x & 31;
  int lrow = lane & 15;
  int rsh  = (lane & 16) ? 8 : 0;

  f32x8 acc[2][4];
  #pragma unroll
  for (int i = 0; i < 2; ++i)
    #pragma unroll
    for (int j = 0; j < 4; ++j)
      acc[i][j] = (f32x8){0.f,0.f,0.f,0.f,0.f,0.f,0.f,0.f};

  gemm_core<D_MODEL>(Y, WT, m0, n0, lane, acc);

  #pragma unroll
  for (int i = 0; i < 2; ++i)
    #pragma unroll
    for (int r = 0; r < 8; ++r) {
      int row = m0 + i * 16 + r + rsh;
      #pragma unroll
      for (int j = 0; j < 4; ++j)
        OUT[(size_t)row * D_MODEL + n0 + j * 16 + lrow] = acc[i][j][r];
    }
}

/* ---------------- host launcher ---------------- */

extern "C" void kernel_launch(void* const* d_in, const int* in_sizes, int n_in,
                              void* d_out, int out_size, void* d_ws, size_t ws_size,
                              hipStream_t stream) {
  const float* x     = (const float*)d_in[0];
  const float* w_qkv = (const float*)d_in[1];
  const float* w_out = (const float*)d_in[2];
  float* out = (float*)d_out;

  bf16_t* p     = (bf16_t*)d_ws;
  bf16_t* xb    = p; p += (size_t)BT * D_MODEL;        /* 16 MB */
  bf16_t* wqkvT = p; p += (size_t)D_MODEL * QKV_N;     /*  6 MB */
  bf16_t* woutT = p; p += (size_t)D_MODEL * D_MODEL;   /*  2 MB */
  bf16_t* q     = p; p += (size_t)BT * D_MODEL;
  bf16_t* kT    = p; p += (size_t)BT * D_MODEL;
  bf16_t* v     = p; p += (size_t)BT * D_MODEL;
  bf16_t* y     = p; p += (size_t)BT * D_MODEL;        /* total ~88 MB */

  const int B256 = 256;
  k_f32_to_bf16<<<(BT * D_MODEL + B256 - 1) / B256, B256, 0, stream>>>(x, xb, BT * D_MODEL);
  k_transpose_to_bf16<<<(QKV_N / 32) * (D_MODEL / 32), B256, 0, stream>>>(w_qkv, wqkvT, QKV_N, D_MODEL);
  k_transpose_to_bf16<<<(D_MODEL / 32) * (D_MODEL / 32), B256, 0, stream>>>(w_out, woutT, D_MODEL, D_MODEL);

  k_qkv_gemm<<<(BT / 32) * (QKV_N / 64) / 8, B256, 0, stream>>>(xb, wqkvT, q, kT, v);
  k_attn<<<(BATCH * N_HEADS * (SEQ / 16)) / 8, B256, 0, stream>>>(q, kT, v, y);
  k_out_gemm<<<(BT / 32) * (D_MODEL / 64) / 8, B256, 0, stream>>>(y, woutT, out);
}
